// MultiheadAttention_with_RoPE_59794534695548
// MI455X (gfx1250) — compile-verified, matrix operand order checked by reading, not run on hardware
//
#include <hip/hip_runtime.h>
#include <hip/hip_bf16.h>

// ---------------------------------------------------------------------------
// MultiheadAttention with RoPE for MI455X (gfx1250), wave32 + WMMA bf16.
//   B=4, S=4096, D_MODEL=1024, N_HEADS=16, D_K=64.  Tokens T = B*S = 16384.
// Pre-pass: x -> bf16, W -> bf16 transposed [n][k]   (memory-bound, tiny)
// Phase 1:  Q/K/V = x@W + b. bf16 WMMA GEMM, f32 accum, RoPE fused epilogue.
//           Staging via GLOBAL_LOAD_ASYNC_TO_LDS_B128 (ASYNCcnt),
//           double-buffered LDS, pipeline manually unrolled x2 so the
//           buffer index is compile-time constant.
// Phase 2:  per-token 16x16 (heads x heads) attention, 6 WMMAs per token.
// ---------------------------------------------------------------------------

typedef __bf16 bf16x16 __attribute__((ext_vector_type(16)));
typedef float  f32x8   __attribute__((ext_vector_type(8)));

#define T_TOKENS 16384
#define DMODEL   1024
#define SEQ      4096
#define LOG1E4   9.210340371976184f   // ln(10000)

__device__ __forceinline__ unsigned short f2bf(float f) {
    unsigned int u = __builtin_bit_cast(unsigned int, f);
    u += 0x7FFFu + ((u >> 16) & 1u);          // round-to-nearest-even
    return (unsigned short)(u >> 16);
}
__device__ __forceinline__ unsigned int pack2(float a, float b) {
    return (unsigned int)f2bf(a) | ((unsigned int)f2bf(b) << 16);
}

union Frag {
    bf16x16 v;
    unsigned short s[16];
    uint4 q[2];
};

// Async global->LDS copy of 16 bytes (per lane). ldsoff = LDS byte address,
// gaddr = 64-bit global address.  Tracked by ASYNCcnt.  No memory clobber:
// ordering vs LDS reads is enforced by s_wait_asynccnt + barriers.
__device__ __forceinline__ void async_b128(unsigned ldsoff, const void* gaddr) {
    asm volatile("global_load_async_to_lds_b128 %0, %1, off"
                 :: "v"(ldsoff), "v"((unsigned long long)gaddr));
}
__device__ __forceinline__ void wait_async0() {
    asm volatile("s_wait_asynccnt 0x0" ::: "memory");
}

// ---------------------------------------------------------------------------
// Pre-pass A: x (f32) -> xb (bf16).  8 elements per thread.
// ---------------------------------------------------------------------------
__global__ __launch_bounds__(256)
void cvt_x_kernel(const float* __restrict__ x, unsigned int* __restrict__ xb) {
    size_t i8 = ((size_t)blockIdx.x * 256 + threadIdx.x) * 8;   // element base
    float4 a = *(const float4*)(x + i8);
    float4 b = *(const float4*)(x + i8 + 4);
    uint4 o;
    o.x = pack2(a.x, a.y); o.y = pack2(a.z, a.w);
    o.z = pack2(b.x, b.y); o.w = pack2(b.z, b.w);
    *(uint4*)(xb + i8 / 2) = o;
}

// ---------------------------------------------------------------------------
// Pre-pass B: W[k][n] (f32) -> WT[n][k] (bf16), 64x64 tiles via LDS.
// grid = (k-tiles=16, n-tiles=16, 3), block = 256.
// ---------------------------------------------------------------------------
__global__ __launch_bounds__(256)
void cvt_w_kernel(const float* __restrict__ Wq, const float* __restrict__ Wk,
                  const float* __restrict__ Wv, unsigned short* __restrict__ WT) {
    __shared__ __align__(16) unsigned short Ts[64 * 80];  // stride 80 -> rows 16B aligned
    const float* W = (blockIdx.z == 0) ? Wq : (blockIdx.z == 1) ? Wk : Wv;
    unsigned short* dst = WT + (size_t)blockIdx.z * DMODEL * DMODEL;

    const int tid = threadIdx.x;
    const int k0 = blockIdx.x * 64;
    const int n0 = blockIdx.y * 64;

    #pragma unroll
    for (int it = 0; it < 4; ++it) {                 // 1024 float4 slots
        int e  = tid + it * 256;
        int k  = e >> 4;                             // 0..63
        int nq = e & 15;
        float4 f = *(const float4*)(W + (size_t)(k0 + k) * DMODEL + n0 + nq * 4);
        Ts[(nq * 4 + 0) * 80 + k] = f2bf(f.x);
        Ts[(nq * 4 + 1) * 80 + k] = f2bf(f.y);
        Ts[(nq * 4 + 2) * 80 + k] = f2bf(f.z);
        Ts[(nq * 4 + 3) * 80 + k] = f2bf(f.w);
    }
    __syncthreads();
    #pragma unroll
    for (int it = 0; it < 2; ++it) {                 // 512 uint4 slots
        int e  = tid + it * 256;
        int n  = e >> 3;                             // 0..63
        int kq = e & 7;
        uint4 v = *(const uint4*)&Ts[n * 80 + kq * 8];
        *(uint4*)(dst + (size_t)(n0 + n) * DMODEL + k0 + kq * 8) = v;
    }
}

// ---------------------------------------------------------------------------
// Phase 1: GEMM  C[16384 x 1024] = xb * WT^T + b  (both operands bf16)
// Block tile 128(M) x 128(N), K-step 32, 256 threads = 8 waves (4 Mw x 2 Nw).
// Each wave: 2 M-subtiles x 4 N-subtiles = 8 WMMAs / K-step.
// ---------------------------------------------------------------------------
#define BM 128
#define BN 128
#define KB 32
#define LS 40   // LDS row stride in bf16 (32 + 8 pad, rows stay 16B aligned)

__device__ __forceinline__
void gemm_step(const unsigned short* XsBuf, const unsigned short* WsBuf,
               int mw, int nw, int r15, int half, f32x8 acc[2][4]) {
    Frag a[2];
    #pragma unroll
    for (int mt = 0; mt < 2; ++mt) {
        const unsigned short* xr = &XsBuf[(mw * 32 + mt * 16 + r15) * LS];
        a[mt].q[0] = *(const uint4*)(xr + half);
        a[mt].q[1] = *(const uint4*)(xr + 16 + half);
    }
    #pragma unroll
    for (int nt = 0; nt < 4; ++nt) {
        Frag b;
        const unsigned short* wr = &WsBuf[(nw * 64 + nt * 16 + r15) * LS];
        b.q[0] = *(const uint4*)(wr + half);
        b.q[1] = *(const uint4*)(wr + 16 + half);
        #pragma unroll
        for (int mt = 0; mt < 2; ++mt)
            acc[mt][nt] = __builtin_amdgcn_wmma_f32_16x16x32_bf16(
                false, a[mt].v, false, b.v, (short)0, acc[mt][nt], false, false);
    }
}

__global__ __launch_bounds__(256)
void qkv_rope_gemm_kernel(const unsigned short* __restrict__ xb,
                          const unsigned short* __restrict__ WT,
                          const float* __restrict__ bias,
                          unsigned short* __restrict__ out,
                          int rope) {
    __shared__ __align__(16) unsigned short Xs[2][BM * LS];   // 2 x 10 KB
    __shared__ __align__(16) unsigned short Ws[2][BN * LS];   // 2 x 10 KB

    const int tid  = threadIdx.x;
    const int lane = tid & 31;
    const int wave = tid >> 5;
    const int r15  = lane & 15;
    const int half = (lane >> 4) * 8;
    const int mw   = wave & 3;          // 0..3 -> 32-row strip
    const int nw   = wave >> 2;         // 0..1 -> 64-col strip

    const int n0 = blockIdx.x * BN;
    const int m0 = blockIdx.y * BM;

    // staging slots: 512 uint4 each for X and W tiles, 2 per thread
    const int srow0 = tid >> 2, skq0 = tid & 3;
    const int srow1 = (tid + 256) >> 2, skq1 = (tid + 256) & 3;

    const unsigned short* xs0 = xb + (size_t)(m0 + srow0) * DMODEL + skq0 * 8;
    const unsigned short* xs1 = xb + (size_t)(m0 + srow1) * DMODEL + skq1 * 8;
    const unsigned short* ws0 = WT + (size_t)(n0 + srow0) * DMODEL + skq0 * 8;
    const unsigned short* ws1 = WT + (size_t)(n0 + srow1) * DMODEL + skq1 * 8;

    const unsigned lx0[2] = { (unsigned)(size_t)&Xs[0][srow0 * LS + skq0 * 8],
                              (unsigned)(size_t)&Xs[1][srow0 * LS + skq0 * 8] };
    const unsigned lx1[2] = { (unsigned)(size_t)&Xs[0][srow1 * LS + skq1 * 8],
                              (unsigned)(size_t)&Xs[1][srow1 * LS + skq1 * 8] };
    const unsigned lw0[2] = { (unsigned)(size_t)&Ws[0][srow0 * LS + skq0 * 8],
                              (unsigned)(size_t)&Ws[1][srow0 * LS + skq0 * 8] };
    const unsigned lw1[2] = { (unsigned)(size_t)&Ws[0][srow1 * LS + skq1 * 8],
                              (unsigned)(size_t)&Ws[1][srow1 * LS + skq1 * 8] };

    auto issue_tile = [&](int buf, int k0) {
        async_b128(lx0[buf], xs0 + k0);
        async_b128(lx1[buf], xs1 + k0);
        async_b128(lw0[buf], ws0 + k0);
        async_b128(lw1[buf], ws1 + k0);
    };

    f32x8 acc[2][4] = {};
    issue_tile(0, 0);

    const int NSTEP = DMODEL / KB;      // 32 (even)
    for (int i = 0; i < NSTEP; i += 2) {
        // ---- even step: compute buf 0, prefetch into buf 1
        wait_async0();
        __syncthreads();
        if (i + 1 < NSTEP) issue_tile(1, (i + 1) * KB);
        gemm_step(Xs[0], Ws[0], mw, nw, r15, half, acc);

        // ---- odd step: compute buf 1, prefetch into buf 0
        wait_async0();
        __syncthreads();
        if (i + 2 < NSTEP) issue_tile(0, (i + 2) * KB);
        gemm_step(Xs[1], Ws[1], mw, nw, r15, half, acc);
    }

    // ---- epilogue: bias (+ RoPE) -> bf16 store
    const float inv_dm = LOG1E4 / (float)DMODEL;
    #pragma unroll
    for (int mt = 0; mt < 2; ++mt) {
        #pragma unroll
        for (int nt = 0; nt < 4; ++nt) {
            int n = n0 + nw * 64 + nt * 16 + r15;
            float bn = bias[n];
            int d = n & 63;                                    // index within head
            float freq = __expf(-(float)(d & ~1) * inv_dm);
            #pragma unroll
            for (int e = 0; e < 8; ++e) {
                int row = m0 + mw * 32 + mt * 16 + half + e;
                float v = acc[mt][nt][e] + bn;
                float p = __shfl_xor(v, 1, 32);                // even<->odd partner
                if (rope) {
                    int pos = row & (SEQ - 1);
                    float ang = (float)pos * freq;
                    float sn, cs;
                    __sincosf(ang, &sn, &cs);
                    v = (n & 1) ? (p * sn + v * cs)            // odd:  xe*sin + xo*cos
                                : (v * cs - p * sn);           // even: xe*cos - xo*sin
                }
                out[(size_t)row * DMODEL + n] = f2bf(v);
            }
        }
    }
}

// ---------------------------------------------------------------------------
// Phase 2: per-token attention. One wave per token; 8 tokens per block.
//   S^T = K_t * Q_t^T  (two 16x16x32 bf16 WMMAs) -> softmax in-lane ->
//   attn accumulator IS a valid B fragment -> out^T = V^T * attn (4 WMMAs).
// ---------------------------------------------------------------------------
__global__ __launch_bounds__(256)
void token_head_attn_kernel(const unsigned short* __restrict__ Qb,
                            const unsigned short* __restrict__ Kb,
                            const unsigned short* __restrict__ Vb,
                            float* __restrict__ out) {
    const int tid  = threadIdx.x;
    const int lane = tid & 31;
    const int wave = tid >> 5;
    const int r15  = lane & 15;
    const int half = (lane >> 4) * 8;

    const int t = blockIdx.x * 8 + wave;

    const unsigned short* qrow = Qb + (size_t)t * DMODEL + r15 * 64;  // head i = r15
    const unsigned short* krow = Kb + (size_t)t * DMODEL + r15 * 64;  // head j = r15
    const unsigned short* vtok = Vb + (size_t)t * DMODEL;

    // ---- S^T[j][i] = sum_d K[j][d] * Q[i][d], d = 0..63 in two K-chunks
    f32x8 st = {};
    #pragma unroll
    for (int c = 0; c < 2; ++c) {
        Frag a, b;
        a.q[0] = *(const uint4*)(krow + c * 32 + half);
        a.q[1] = *(const uint4*)(krow + c * 32 + 16 + half);
        b.q[0] = *(const uint4*)(qrow + c * 32 + half);
        b.q[1] = *(const uint4*)(qrow + c * 32 + 16 + half);
        st = __builtin_amdgcn_wmma_f32_16x16x32_bf16(
            false, a.v, false, b.v, (short)0, st, false, false);
    }

    // ---- softmax over j (rows of S^T): 8 in-lane values + partner half-wave
    float mx = -3.4e38f;
    #pragma unroll
    for (int e = 0; e < 8; ++e) { st[e] *= 0.125f; mx = fmaxf(mx, st[e]); }
    mx = fmaxf(mx, __shfl_xor(mx, 16, 32));
    float sum = 0.f;
    #pragma unroll
    for (int e = 0; e < 8; ++e) { st[e] = __expf(st[e] - mx); sum += st[e]; }
    sum += __shfl_xor(sum, 16, 32);
    float inv = 1.0f / sum;

    // ---- pack attn as B fragment (K = j in [0,16), padded to 32 with zeros)
    Frag pb;
    #pragma unroll
    for (int e = 0; e < 8; ++e) pb.s[e] = f2bf(st[e] * inv);
    #pragma unroll
    for (int e = 8; e < 16; ++e) pb.s[e] = 0;

    // ---- out^T = V^T * attn, d in 4 chunks of 16
    #pragma unroll
    for (int c = 0; c < 4; ++c) {
        Frag av;
        int d = c * 16 + r15;                      // A-matrix row M = d
        #pragma unroll
        for (int e = 0; e < 8; ++e)                // K = j = half + e
            av.s[e] = vtok[(half + e) * 64 + d];
        #pragma unroll
        for (int e = 8; e < 16; ++e) av.s[e] = 0;

        f32x8 o = {};
        o = __builtin_amdgcn_wmma_f32_16x16x32_bf16(
            false, av.v, false, pb.v, (short)0, o, false, false);

        // D rows are d = c*16 + half + e, N = head i = r15 -> contiguous floats
        float* orow = out + (size_t)t * DMODEL + r15 * 64 + c * 16 + half;
        float4 s0 = make_float4(o[0], o[1], o[2], o[3]);
        float4 s1 = make_float4(o[4], o[5], o[6], o[7]);
        *(float4*)(orow)     = s0;
        *(float4*)(orow + 4) = s1;
    }
}

// ---------------------------------------------------------------------------
extern "C" void kernel_launch(void* const* d_in, const int* in_sizes, int n_in,
                              void* d_out, int out_size, void* d_ws, size_t ws_size,
                              hipStream_t stream) {
    const float* x  = (const float*)d_in[0];
    const float* Wq = (const float*)d_in[1];
    const float* bq = (const float*)d_in[2];
    const float* Wk = (const float*)d_in[3];
    const float* bk = (const float*)d_in[4];
    const float* Wv = (const float*)d_in[5];
    const float* bv = (const float*)d_in[6];
    float* out = (float*)d_out;

    const size_t mat  = (size_t)T_TOKENS * DMODEL;    // 16.7M elements
    const size_t wmat = (size_t)DMODEL * DMODEL;      // 1M elements

    unsigned short* xbuf = (unsigned short*)d_ws;     // bf16 x
    unsigned short* WT   = xbuf + mat;                // 3 transposed bf16 weights
    unsigned short* Qb   = WT + 3 * wmat;
    unsigned short* Kb   = Qb + mat;
    unsigned short* Vb   = Kb + mat;

    cvt_x_kernel<<<mat / (256 * 8), 256, 0, stream>>>(x, (unsigned int*)xbuf);
    cvt_w_kernel<<<dim3(16, 16, 3), 256, 0, stream>>>(Wq, Wk, Wv, WT);

    dim3 gg(DMODEL / BN, T_TOKENS / BM, 1);           // 8 x 128 blocks
    qkv_rope_gemm_kernel<<<gg, 256, 0, stream>>>(xbuf, WT + 0 * wmat, bq, Qb, 1);
    qkv_rope_gemm_kernel<<<gg, 256, 0, stream>>>(xbuf, WT + 1 * wmat, bk, Kb, 1);
    qkv_rope_gemm_kernel<<<gg, 256, 0, stream>>>(xbuf, WT + 2 * wmat, bv, Vb, 0);

    token_head_attn_kernel<<<T_TOKENS / 8, 256, 0, stream>>>(Qb, Kb, Vb, out);
}
